// Head_76209899700400
// MI455X (gfx1250) — matrix operand order repaired and checked by reading, not compiled, not run on hardware
//
#include <hip/hip_runtime.h>
#include <hip/hip_bf16.h>

#define N_EMBD 1024
#define HEAD   128
#define SEQ    2048
#define BATCH  8

typedef __attribute__((ext_vector_type(16))) __bf16       v16bf;
typedef __attribute__((ext_vector_type(8)))  float        v8f;
typedef __attribute__((ext_vector_type(8)))  unsigned int v8u;
typedef __attribute__((ext_vector_type(4)))  int          v4i_;

// ---------------- CDNA5 async global->LDS copy (guarded) ----------------
#if __has_builtin(__builtin_amdgcn_global_load_async_to_lds_b128) && \
    __has_builtin(__builtin_amdgcn_s_wait_asynccnt)
#define USE_ASYNC_LDS 1
#endif

static __device__ __forceinline__ void cp16_to_lds(void* lds, const void* g) {
#if defined(USE_ASYNC_LDS)
  // signature (from probe diagnostic): (AS1 v4i* gsrc, AS3 v4i* ldst, imm, imm)
  __builtin_amdgcn_global_load_async_to_lds_b128(
      (__attribute__((address_space(1))) v4i_*)g,
      (__attribute__((address_space(3))) v4i_*)lds, 0, 0);
#else
  *(uint4*)lds = *(const uint4*)g;
#endif
}
static __device__ __forceinline__ void async_wait_all() {
#if defined(USE_ASYNC_LDS)
  __builtin_amdgcn_s_wait_asynccnt(0);
#endif
}

// ---------------- bf16 helpers ----------------
static __device__ __forceinline__ unsigned short f2bf(float f) {
  unsigned u = __float_as_uint(f);
  u += 0x7FFFu + ((u >> 16) & 1u);
  return (unsigned short)(u >> 16);
}
#if __has_builtin(__builtin_amdgcn_cvt_pk_bf16_f32)
typedef __attribute__((ext_vector_type(2))) __bf16 v2bf;
static __device__ __forceinline__ unsigned int pkbf(float lo, float hi) {
  return __builtin_bit_cast(unsigned int, __builtin_amdgcn_cvt_pk_bf16_f32(lo, hi));
}
#else
static __device__ __forceinline__ unsigned int pkbf(float lo, float hi) {
  return (unsigned int)f2bf(lo) | ((unsigned int)f2bf(hi) << 16);
}
#endif
static __device__ __forceinline__ v8f wmma_bf16(v8u a, v8u b, v8f c) {
  return __builtin_amdgcn_wmma_f32_16x16x32_bf16(
      false, __builtin_bit_cast(v16bf, a),
      false, __builtin_bit_cast(v16bf, b),
      (short)0, c, false, false);
}

// ========== Kernel 0: one-time fp32 -> bf16 weight conversion ==========
__global__ __launch_bounds__(256) void wcvt(
    const float* __restrict__ Wq, const float* __restrict__ Wk,
    const float* __restrict__ Wv,
    unsigned int* __restrict__ oq, unsigned int* __restrict__ ok,
    unsigned int* __restrict__ ov)
{
  const float* src = blockIdx.y == 0 ? Wq : (blockIdx.y == 1 ? Wk : Wv);
  unsigned int* dst = blockIdx.y == 0 ? oq : (blockIdx.y == 1 ? ok : ov);
  const size_t i = ((size_t)blockIdx.x * 256 + threadIdx.x) * 8;
  float4 f0 = *(const float4*)(src + i);
  float4 f1 = *(const float4*)(src + i + 4);
  uint4 o;
  o.x = pkbf(f0.x, f0.y); o.y = pkbf(f0.z, f0.w);
  o.z = pkbf(f1.x, f1.y); o.w = pkbf(f1.z, f1.w);
  *(uint4*)(dst + i / 2) = o;
}

// ========== Kernel 1: fused QKV projection (bf16 WMMA) ==========
// Block = 64 rows x 128 cols; 8 waves each own 16 cols and sweep 4 M-tiles,
// so every weight B-fragment feeds 4 WMMAs. x tile staged once through LDS.
__global__ __launch_bounds__(256) void qkv_wmma(
    const float* __restrict__ x,
    const unsigned short* __restrict__ wq, const unsigned short* __restrict__ wk,
    const unsigned short* __restrict__ wv,
    unsigned short* __restrict__ qo, unsigned short* __restrict__ ko,
    unsigned short* __restrict__ vt)
{
  __shared__ __align__(16) unsigned int Xt[2][64 * 16];  // 64 rows x 32 bf16

  const int tid  = threadIdx.x;
  const int lane = tid & 31;
  const int wave = tid >> 5;
  const int n    = lane & 15;
  const int half = lane >> 4;
  const int mblock = blockIdx.x * 64;
  const int hbase  = wave * 16;

  const int srow = tid >> 2;        // staging: row 0..63
  const int scol = (tid & 3) * 8;   // staging: col 0,8,16,24
  const float* xs = x + (size_t)(mblock + srow) * N_EMBD + scol;

  const unsigned short* wqr = wq + (size_t)(hbase + n) * N_EMBD;
  const unsigned short* wkr = wk + (size_t)(hbase + n) * N_EMBD;
  const unsigned short* wvr = wv + (size_t)(hbase + n) * N_EMBD;

  v8f zero = {};
  v8f acc[3][4];
#pragma unroll
  for (int j = 0; j < 3; ++j)
#pragma unroll
    for (int mt = 0; mt < 4; ++mt) acc[j][mt] = zero;

  auto stage = [&](int buf, int k0) {
    float4 f0 = *(const float4*)(xs + k0);
    float4 f1 = *(const float4*)(xs + k0 + 4);
    uint4 o;
    o.x = pkbf(f0.x, f0.y); o.y = pkbf(f0.z, f0.w);
    o.z = pkbf(f1.x, f1.y); o.w = pkbf(f1.z, f1.w);
    *(uint4*)&Xt[buf][srow * 16 + scol / 2] = o;
  };

  stage(0, 0);
  for (int ks = 0; ks < N_EMBD / 32; ++ks) {
    __syncthreads();
    const int cur = ks & 1;
    if (ks + 1 < N_EMBD / 32) stage(cur ^ 1, (ks + 1) * 32);
    const int k0 = ks * 32;
    __builtin_prefetch(wqr + k0 + 64, 0, 0);

    v8u bq, bk, bv;   // B elems 0..15 : K = half*16 + e (contiguous bf16)
#pragma unroll
    for (int u = 0; u < 8; ++u) {
      bq[u] = *(const unsigned int*)(wqr + k0 + half * 16 + 2 * u);
      bk[u] = *(const unsigned int*)(wkr + k0 + half * 16 + 2 * u);
      bv[u] = *(const unsigned int*)(wvr + k0 + half * 16 + 2 * u);
    }
#pragma unroll
    for (int mt = 0; mt < 4; ++mt) {
      const unsigned int* xrow = &Xt[cur][(mt * 16 + n) * 16];
      v8u a;  // A dword u: K pair = half*16 + 2u (contiguous)
#pragma unroll
      for (int u = 0; u < 4; ++u) {
        a[u]     = xrow[half * 8 + u];
        a[4 + u] = xrow[half * 8 + 4 + u];
      }
      acc[0][mt] = wmma_bf16(a, bq, acc[0][mt]);
      acc[1][mt] = wmma_bf16(a, bk, acc[1][mt]);
      acc[2][mt] = wmma_bf16(a, bv, acc[2][mt]);
    }
  }

  const int b = mblock / SEQ;
  const int t = mblock - b * SEQ;
#pragma unroll
  for (int mt = 0; mt < 4; ++mt) {
    const int h = hbase + n;
#pragma unroll
    for (int r = 0; r < 8; ++r) {   // D: VGPR r -> M = r + 8*half, N = n
      const int lm = mt * 16 + r + 8 * half;
      qo[(size_t)(mblock + lm) * HEAD + h] = f2bf(acc[0][mt][r]);
      ko[(size_t)(mblock + lm) * HEAD + h] = f2bf(acc[1][mt][r]);
    }
    // v stored transposed [B,H,T]; consecutive r -> consecutive t: pack dwords
    unsigned int* vrow =
        (unsigned int*)(vt + ((size_t)b * HEAD + h) * SEQ + t + mt * 16 + 8 * half);
#pragma unroll
    for (int r = 0; r < 8; r += 2)
      vrow[r / 2] = pkbf(acc[2][mt][r], acc[2][mt][r + 1]);
  }
}

// ========== Kernel 2: causal flash attention (bf16 WMMA) ==========
// 8 consecutive q-tiles (same batch) per block; K/V blocks staged cooperatively
// into LDS with double buffering + async global->LDS copies.
__global__ __launch_bounds__(256) void attn_wmma(
    const unsigned short* __restrict__ qm,
    const unsigned short* __restrict__ km,
    const unsigned short* __restrict__ vt,
    float* __restrict__ out)
{
  __shared__ __align__(16) unsigned short Kt[2][32 * HEAD];   // [key][head]
  __shared__ __align__(16) unsigned short Vt[2][HEAD * 32];   // [head][key]
  __shared__ __align__(16) unsigned short ldsP[8][16 * 32];   // per-wave P

  const int tid  = threadIdx.x;
  const int lane = tid & 31;
  const int wave = tid >> 5;
  const int n    = lane & 15;
  const int half = lane >> 4;

  const int qt    = blockIdx.x * 8 + wave;
  const int mbase = qt * 16;
  const int b     = mbase / SEQ;
  const int lrow  = mbase - b * SEQ;
  const int lrow_blk = (blockIdx.x % (SEQ / 128)) * 128;

  const unsigned short* kb = km + (size_t)b * SEQ * HEAD;
  const unsigned short* vb = vt + (size_t)b * HEAD * SEQ;

  // Q as A-fragments over the 128-wide head dim
  v8u aq[4];
  const unsigned short* qr = qm + (size_t)(mbase + n) * HEAD;
#pragma unroll
  for (int f = 0; f < 4; ++f) {
#pragma unroll
    for (int u = 0; u < 4; ++u) {
      aq[f][u]     = *(const unsigned int*)(qr + f * 32 + half * 16 + 2 * u);
      aq[f][4 + u] = *(const unsigned int*)(qr + f * 32 + half * 16 + 8 + 2 * u);
    }
  }

  float mst[8], lst[8];
#pragma unroll
  for (int r = 0; r < 8; ++r) { mst[r] = -__builtin_inff(); lst[r] = 0.f; }
  v8f zero = {};
  v8f o[8];
#pragma unroll
  for (int f = 0; f < 8; ++f) o[f] = zero;

  const float scale = 0.08838834764831845f;     // 1/sqrt(128)
  const int my_nkb  = (lrow + 16 + 31) / 32;    // this wave's causal extent
  const int nkb_max = lrow_blk / 32 + 4;        // block-wide extent

  auto stageKV = [&](int buf, int s0) {
#pragma unroll
    for (int i = 0; i < 2; ++i) {
      const int c = tid + 256 * i;              // 512 x 16B chunks each
      const int krow = c >> 4, koff = (c & 15) * 8;
      cp16_to_lds(&Kt[buf][krow * HEAD + koff],
                  kb + (size_t)(s0 + krow) * HEAD + koff);
      const int hrow = c >> 2, voff = (c & 3) * 8;
      cp16_to_lds(&Vt[buf][hrow * 32 + voff],
                  vb + (size_t)hrow * SEQ + s0 + voff);
    }
  };

  stageKV(0, 0);
  for (int sb = 0; sb < nkb_max; ++sb) {
    async_wait_all();
    __syncthreads();                   // staged buffer visible to all waves
    const int cur = sb & 1;
    if (sb + 1 < nkb_max) stageKV(cur ^ 1, (sb + 1) * 32);

    if (sb < my_nkb) {                 // wave-uniform: EXEC stays all-ones
      const int s0 = sb * 32;

      // S = Q * K^T (two 16x16 tiles over 32 keys)
      v8f sa0 = zero, sa1 = zero;
#pragma unroll
      for (int f = 0; f < 4; ++f) {
        const unsigned short* kr0 = &Kt[cur][n * HEAD + f * 32 + half * 16];
        const unsigned short* kr1 = &Kt[cur][(16 + n) * HEAD + f * 32 + half * 16];
        v8u bk0, bk1;
#pragma unroll
        for (int u = 0; u < 8; ++u) {
          bk0[u] = *(const unsigned int*)(kr0 + 2 * u);
          bk1[u] = *(const unsigned int*)(kr1 + 2 * u);
        }
        sa0 = wmma_bf16(aq[f], bk0, sa0);
        sa1 = wmma_bf16(aq[f], bk1, sa1);
      }

      // online softmax (row m = r + 8*half lives in one 16-lane half)
      unsigned short* lp = &ldsP[wave][0];
#pragma unroll
      for (int r = 0; r < 8; ++r) {
        const int row = lrow + r + 8 * half;
        float s0v = (s0 + n)      <= row ? sa0[r] * scale : -__builtin_inff();
        float s1v = (s0 + 16 + n) <= row ? sa1[r] * scale : -__builtin_inff();
        float mx = fmaxf(s0v, s1v);
#pragma unroll
        for (int d = 1; d < 16; d <<= 1) mx = fmaxf(mx, __shfl_xor(mx, d, 32));
        const float mnew  = fmaxf(mst[r], mx);
        const float alpha = __expf(mst[r] - mnew);
        const float p0 = __expf(s0v - mnew);
        const float p1 = __expf(s1v - mnew);
        float rs = p0 + p1;
#pragma unroll
        for (int d = 1; d < 16; d <<= 1) rs += __shfl_xor(rs, d, 32);
        lst[r] = lst[r] * alpha + rs;
        mst[r] = mnew;
#pragma unroll
        for (int f = 0; f < 8; ++f) o[f][r] *= alpha;
        lp[(r + 8 * half) * 32 + n]      = f2bf(p0);
        lp[(r + 8 * half) * 32 + 16 + n] = f2bf(p1);
      }

      // P re-read in A layout (same-wave DS ops are in-order)
      v8u pa;
      const unsigned short* lrp = &ldsP[wave][n * 32];
#pragma unroll
      for (int u = 0; u < 4; ++u) {
        pa[u]     = *(const unsigned int*)(lrp + half * 16 + 2 * u);
        pa[4 + u] = *(const unsigned int*)(lrp + half * 16 + 8 + 2 * u);
      }

      // O += P * V  (V^T tile: K-runs contiguous in LDS)
#pragma unroll
      for (int f = 0; f < 8; ++f) {
        const unsigned short* vr = &Vt[cur][(f * 16 + n) * 32 + half * 16];
        v8u bvv;
#pragma unroll
        for (int u = 0; u < 8; ++u) bvv[u] = *(const unsigned int*)(vr + 2 * u);
        o[f] = wmma_bf16(pa, bvv, o[f]);
      }
    }
  }

  // normalize, store fp32 [B,T,H]
#pragma unroll
  for (int f = 0; f < 8; ++f) {
#pragma unroll
    for (int r = 0; r < 8; ++r) {
      out[(size_t)(mbase + r + 8 * half) * HEAD + f * 16 + n] = o[f][r] / lst[r];
    }
  }
}

extern "C" void kernel_launch(void* const* d_in, const int* in_sizes, int n_in,
                              void* d_out, int out_size, void* d_ws, size_t ws_size,
                              hipStream_t stream) {
  const float* x  = (const float*)d_in[0];
  const float* Wq = (const float*)d_in[1];
  const float* Wk = (const float*)d_in[2];
  const float* Wv = (const float*)d_in[3];
  float* out = (float*)d_out;

  const size_t nQKV = (size_t)BATCH * SEQ * HEAD;   // 2M bf16 elems each
  unsigned short* qw  = (unsigned short*)d_ws;
  unsigned short* kw  = qw + nQKV;
  unsigned short* vw  = kw + nQKV;                  // transposed [B,H,T]
  unsigned short* wqb = vw + nQKV;                  // bf16 weights
  unsigned short* wkb = wqb + (size_t)HEAD * N_EMBD;
  unsigned short* wvb = wkb + (size_t)HEAD * N_EMBD;

  wcvt<<<dim3(HEAD * N_EMBD / 2048, 3), dim3(256), 0, stream>>>(
      Wq, Wk, Wv, (unsigned int*)wqb, (unsigned int*)wkb, (unsigned int*)wvb);
  qkv_wmma<<<dim3(BATCH * SEQ / 64), dim3(256), 0, stream>>>(
      x, wqb, wkb, wvb, qw, kw, vw);
  attn_wmma<<<dim3(BATCH * SEQ / 16 / 8), dim3(256), 0, stream>>>(qw, kw, vw, out);
}
